// DECConsLoss_12283606468279
// MI455X (gfx1250) — matrix operand order repaired and checked by reading, not compiled, not run on hardware
//
#include <hip/hip_runtime.h>
#include <hip/hip_bf16.h>

typedef __attribute__((ext_vector_type(16))) _Float16 v16h;
typedef __attribute__((ext_vector_type(4)))  _Float16 v4h;
typedef __attribute__((ext_vector_type(8)))  float    v8f;
typedef __attribute__((ext_vector_type(4)))  int      i4v;

#define TOK     65536          // B*T*N = 8*8*1024 total tokens
#define CDIM    512
#define GRPS    32             // G
#define MTOK    2048           // M tokens per group
#define SLOTS   16
#define INV_T   10.0f          // 1/TEMPERATURE
#define LSCALE  (-1.4285714285714286f)  // -(T/baseT)
#define LDS_STRIDE 80          // 64 data bytes + 16 pad, keeps b128 alignment
#define TILE_BYTES (128 * LDS_STRIDE)   // one A or B tile region

union F16Frag { v16h v; float4 f4[2]; };

// ---------------------------------------------------------------------------
// CDNA5 async global->LDS staging (ASYNCcnt-tracked DMA), with safe fallback.
// Builtin signature (from hipcc diagnostic): (int4 as1*, int4 as3*, Imm, Imm)
// ---------------------------------------------------------------------------
#if __has_builtin(__builtin_amdgcn_global_load_async_to_lds_b128)
#define HAVE_ASYNC_LDS 1
typedef __attribute__((address_space(1))) i4v g_i4;
typedef __attribute__((address_space(3))) i4v l_i4;
#if __has_builtin(__builtin_amdgcn_s_wait_asynccnt)
#define WAIT_ASYNC(n) __builtin_amdgcn_s_wait_asynccnt(n)
#else
#define WAIT_ASYNC(n) asm volatile("s_wait_asynccnt %0" ::"i"(n) : "memory")
#endif
#else
#define HAVE_ASYNC_LDS 0
#define WAIT_ASYNC(n) do {} while (0)
#endif

static __device__ __forceinline__ void stage_b128(const _Float16* g, char* l) {
#if HAVE_ASYNC_LDS
  __builtin_amdgcn_global_load_async_to_lds_b128((g_i4*)g, (l_i4*)l, 0, 0);
#else
  *(float4*)l = *(const float4*)g;
#endif
}

// ---------------------------------------------------------------------------
// Kernel 1: L2-normalize each 512-float row, emit f16. One wave per row.
// ---------------------------------------------------------------------------
__global__ void norm_rows_kernel(const float* __restrict__ in,
                                 _Float16* __restrict__ out) {
  const int wid  = (blockIdx.x * blockDim.x + threadIdx.x) >> 5;  // row id
  const int lane = threadIdx.x & 31;
  const float* row = in + (size_t)wid * CDIM;
  float4 v[4];
  float ss = 0.f;
#pragma unroll
  for (int q = 0; q < 4; ++q) {
    v[q] = *(const float4*)(row + q * 128 + lane * 4);
    ss = fmaf(v[q].x, v[q].x, ss);
    ss = fmaf(v[q].y, v[q].y, ss);
    ss = fmaf(v[q].z, v[q].z, ss);
    ss = fmaf(v[q].w, v[q].w, ss);
  }
#pragma unroll
  for (int off = 16; off > 0; off >>= 1) ss += __shfl_xor(ss, off, 32);
  const float inv = 1.0f / fmaxf(sqrtf(ss), 1e-12f);
  _Float16* orow = out + (size_t)wid * CDIM;
#pragma unroll
  for (int q = 0; q < 4; ++q) {
    v4h hv;
    hv[0] = (_Float16)(v[q].x * inv);
    hv[1] = (_Float16)(v[q].y * inv);
    hv[2] = (_Float16)(v[q].z * inv);
    hv[3] = (_Float16)(v[q].w * inv);
    *(v4h*)(orow + q * 128 + lane * 4) = hv;
  }
}

// ---------------------------------------------------------------------------
// Kernel 2: slot id = argmax over S=16 of grp_masks[bt, s, n]  (first max)
// ---------------------------------------------------------------------------
__global__ void grp_argmax_kernel(const float* __restrict__ gm,
                                  int* __restrict__ grp) {
  const int t  = blockIdx.x * blockDim.x + threadIdx.x;  // token 0..65535
  const int bt = t >> 10;
  const int nn = t & 1023;
  const float* p = gm + (size_t)bt * SLOTS * 1024 + nn;
  float best = p[0];
  int bi = 0;
#pragma unroll
  for (int s = 1; s < SLOTS; ++s) {
    const float vv = p[(size_t)s * 1024];
    if (vv > best) { best = vv; bi = s; }
  }
  grp[t] = bi;
}

// ---------------------------------------------------------------------------
// Kernel 3: fused GEMM (WMMA f16 -> f32) + online softmax-loss reduction.
// Block: 256 thr = 8 waves; 128 rows x 128 cols per col-tile, K chunks of 32.
// Double-buffered LDS; async DMA (stage k+1) overlaps WMMA compute (stage k).
// ---------------------------------------------------------------------------
__global__ void __launch_bounds__(256, 2)
cons_loss_main(const _Float16* __restrict__ ft, const _Float16* __restrict__ fc,
               const int* __restrict__ grp, float* __restrict__ partials) {
  __shared__ float4 ldsbuf[4 * 128 * (LDS_STRIDE / 16)];  // 2 x (A,B) = 40KB
  __shared__ float  wsum[8];
  char* const lds0 = (char*)ldsbuf;

  const int g    = blockIdx.y;         // group 0..31
  const int rt   = blockIdx.x;         // row block 0..15
  const int tid  = threadIdx.x;
  const int w    = tid >> 5;           // wave 0..7 (owns rows w*16..+15)
  const int lane = tid & 31;
  const int n    = lane & 15;          // C/D column within 16x16 tile
  const int h    = lane >> 4;          // half-wave

  const int    gbase   = g * MTOK;
  const int    rowloc0 = rt * 128;                   // local row base
  const size_t rowtok0 = (size_t)gbase + rowloc0;    // global token base

  int grpm[8], mloc[8];
#pragma unroll
  for (int r = 0; r < 8; ++r) {
    mloc[r] = rowloc0 + w * 16 + r + 8 * h;          // local row in [0,2048)
    grpm[r] = grp[gbase + mloc[r]];
  }

  float mx[8], sm[8], ps[8], pc[8], dg[8];
#pragma unroll
  for (int r = 0; r < 8; ++r) {
    mx[r] = -__builtin_inff(); sm[r] = 0.f; ps[r] = 0.f; pc[r] = 0.f; dg[r] = 0.f;
  }

  const int srow = tid >> 1;           // staging row 0..127
  const int sq   = (tid & 1) * 32;     // byte offset in row: 0 or 32

  for (int jt = 0; jt < MTOK; jt += 128) {           // 16 col tiles
    v8f acc[8];
#pragma unroll
    for (int t = 0; t < 8; ++t) acc[t] = (v8f)0.0f;

    const size_t coltok0 = (size_t)gbase + jt;

    // ---- stage K-chunk 0 into buffer 0 (4 b128 transfers per thread) ----
    {
      const _Float16* ga = ft + (rowtok0 + srow) * CDIM + (sq >> 1);
      const _Float16* gb = fc + (coltok0 + srow) * CDIM + (sq >> 1);
      char* lA = lds0 + srow * LDS_STRIDE + sq;
      char* lB = lA + TILE_BYTES;
      stage_b128(ga,     lA);
      stage_b128(ga + 8, lA + 16);
      stage_b128(gb,     lB);
      stage_b128(gb + 8, lB + 16);
    }

    for (int kt = 0; kt < 16; ++kt) {                // 16 K chunks of 32
      const int  cur  = kt & 1;
      char* const curA = lds0 + cur * 2 * TILE_BYTES;
      char* const curB = curA + TILE_BYTES;

      if (kt < 15) {
        // issue stage kt+1 into the other buffer; its previous readers all
        // passed the trailing barrier of iteration kt-1 -> no WAR hazard.
        const int kk = (kt + 1) * 32;
        const _Float16* ga = ft + (rowtok0 + srow) * CDIM + kk + (sq >> 1);
        const _Float16* gb = fc + (coltok0 + srow) * CDIM + kk + (sq >> 1);
        char* lA = lds0 + (cur ^ 1) * 2 * TILE_BYTES + srow * LDS_STRIDE + sq;
        char* lB = lA + TILE_BYTES;
        stage_b128(ga,     lA);
        stage_b128(ga + 8, lA + 16);
        stage_b128(gb,     lB);
        stage_b128(gb + 8, lB + 16);
#if !HAVE_ASYNC_LDS
        __builtin_prefetch(ga + 32, 0, 1);
        __builtin_prefetch(gb + 32, 0, 1);
#endif
        WAIT_ASYNC(4);   // oldest 4 (stage kt) done; stage kt+1 stays in flight
      } else {
        WAIT_ASYNC(0);   // drain last stage
      }
      __syncthreads();   // stage kt LDS writes visible to every wave

      // A fragment: lane holds row m=lane&15; 16 contiguous K starting at h*16
      F16Frag fa;
      const char* arow = curA + (w * 16 + n) * LDS_STRIDE;
      fa.f4[0] = *(const float4*)(arow + (h << 5));
      fa.f4[1] = *(const float4*)(arow + 16 + (h << 5));

#pragma unroll
      for (int t = 0; t < 8; ++t) {
        // B fragment: lane holds col n; 16 contiguous K starting at h*16
        F16Frag fb;
        const char* brow = curB + (t * 16 + n) * LDS_STRIDE;
        fb.f4[0] = *(const float4*)(brow + (h << 5));
        fb.f4[1] = *(const float4*)(brow + 16 + (h << 5));
        acc[t] = __builtin_amdgcn_wmma_f32_16x16x32_f16(
            false, fa.v, false, fb.v, (short)0, acc[t], false, false);
      }
      __syncthreads();   // all readers of buffer `cur` done -> reusable
    }

    // ---- epilogue for this 16x128 wave tile: online softmax state update ----
#pragma unroll
    for (int t = 0; t < 8; ++t)
#pragma unroll
      for (int r = 0; r < 8; ++r) acc[t][r] *= INV_T;   // fold 1/TEMPERATURE

    float tmax[8];
#pragma unroll
    for (int r = 0; r < 8; ++r) tmax[r] = -__builtin_inff();
#pragma unroll
    for (int t = 0; t < 8; ++t)
#pragma unroll
      for (int r = 0; r < 8; ++r) tmax[r] = fmaxf(tmax[r], acc[t][r]);
#pragma unroll
    for (int r = 0; r < 8; ++r) {
#pragma unroll
      for (int off = 1; off < 16; off <<= 1)
        tmax[r] = fmaxf(tmax[r], __shfl_xor(tmax[r], off, 32));
      const float nm = fmaxf(mx[r], tmax[r]);
      sm[r] *= __expf(mx[r] - nm);
      mx[r] = nm;
    }

    float es[8], pls[8], plc[8];
#pragma unroll
    for (int r = 0; r < 8; ++r) { es[r] = 0.f; pls[r] = 0.f; plc[r] = 0.f; }
#pragma unroll
    for (int t = 0; t < 8; ++t) {
      const int jl = jt + t * 16 + n;                   // local col in [0,2048)
      const int gj = grp[gbase + jl];
#pragma unroll
      for (int r = 0; r < 8; ++r) {
        const float vv = acc[t][r];
        es[r] += __expf(vv - mx[r]);
        const bool p = (gj == grpm[r]);                 // diag is always same-slot
        pls[r] += p ? vv : 0.f;
        plc[r] += p ? 1.f : 0.f;
        dg[r] = (jl == mloc[r]) ? vv : dg[r];
      }
    }
#pragma unroll
    for (int r = 0; r < 8; ++r) {
#pragma unroll
      for (int off = 1; off < 16; off <<= 1) {
        es[r]  += __shfl_xor(es[r],  off, 32);
        pls[r] += __shfl_xor(pls[r], off, 32);
        plc[r] += __shfl_xor(plc[r], off, 32);
      }
      sm[r] += es[r]; ps[r] += pls[r]; pc[r] += plc[r];
    }
  }

  // gather diagonal across the 16-lane group (exactly one lane captured it)
#pragma unroll
  for (int r = 0; r < 8; ++r)
#pragma unroll
    for (int off = 1; off < 16; off <<= 1) dg[r] += __shfl_xor(dg[r], off, 32);

  // per-row loss; lane with n==r owns row r+8h -> each of 16 rows counted once
  float contrib = 0.f;
#pragma unroll
  for (int r = 0; r < 8; ++r) {
    const float lse  = mx[r] + __logf(sm[r]);
    const float semi = LSCALE * ((ps[r] - pc[r] * lse) / pc[r]);
    const float posl = LSCALE * (dg[r] - lse);
    contrib += (n == r) ? (semi + posl) : 0.f;
  }
#pragma unroll
  for (int off = 16; off > 0; off >>= 1) contrib += __shfl_xor(contrib, off, 32);

  if (lane == 0) wsum[w] = contrib;
  __syncthreads();
  if (tid == 0) {
    float tot = 0.f;
#pragma unroll
    for (int i = 0; i < 8; ++i) tot += wsum[i];
    partials[blockIdx.y * 16 + blockIdx.x] = tot;
  }
}

// ---------------------------------------------------------------------------
// Kernel 4: deterministic final reduction of 512 partials -> scalar output
// ---------------------------------------------------------------------------
__global__ void finalize_kernel(const float* __restrict__ partials,
                                float* __restrict__ out) {
  __shared__ float sh[256];
  const int tid = threadIdx.x;
  sh[tid] = partials[tid] + partials[tid + 256];
  __syncthreads();
  for (int o = 128; o > 0; o >>= 1) {
    if (tid < o) sh[tid] += sh[tid + o];
    __syncthreads();
  }
  if (tid == 0) out[0] = sh[0] / (float)TOK * 0.5f;   // mean(...)/2
}

// ---------------------------------------------------------------------------
extern "C" void kernel_launch(void* const* d_in, const int* in_sizes, int n_in,
                              void* d_out, int out_size, void* d_ws, size_t ws_size,
                              hipStream_t stream) {
  const float* feat_tr = (const float*)d_in[0];
  const float* feat_cr = (const float*)d_in[1];
  const float* gmasks  = (const float*)d_in[2];
  float* out = (float*)d_out;

  _Float16* ft16 = (_Float16*)d_ws;
  _Float16* fc16 = ft16 + (size_t)TOK * CDIM;
  int*   grp      = (int*)((char*)d_ws + 2 * (size_t)TOK * CDIM * sizeof(_Float16));
  float* partials = (float*)((char*)grp + (size_t)TOK * sizeof(int));

  norm_rows_kernel<<<TOK / 8, 256, 0, stream>>>(feat_tr, ft16);
  norm_rows_kernel<<<TOK / 8, 256, 0, stream>>>(feat_cr, fc16);
  grp_argmax_kernel<<<TOK / 256, 256, 0, stream>>>(gmasks, grp);

  dim3 grid(MTOK / 128, GRPS);   // (16, 32)
  cons_loss_main<<<grid, 256, 0, stream>>>(ft16, fc16, grp, partials);

  finalize_kernel<<<1, 256, 0, stream>>>(partials, out);
}
